// HeteroGat_30442728194313
// MI455X (gfx1250) — matrix operand order repaired and checked by reading, not compiled, not run on hardware
//
#include <hip/hip_runtime.h>
#include <hip/hip_bf16.h>

#define NN 50000   // nodes
#define EE 600000  // edges
#define HH 4       // heads
#define DF 128     // D_IN == D_OUT == H*C

typedef __attribute__((ext_vector_type(16))) __bf16 v16bf;
typedef __attribute__((ext_vector_type(8)))  float  v8f;
typedef __attribute__((ext_vector_type(4)))  float  v4f;

__device__ __forceinline__ unsigned short f2bf(float f) {
  union { float f; unsigned u; } c; c.f = f;
  unsigned u = c.u;
  u += 0x7FFFu + ((u >> 16) & 1u);   // round-to-nearest-even
  return (unsigned short)(u >> 16);
}

__device__ __forceinline__ void atomicMaxF(float* addr, float val) {
  // sign-aware max on float bits; init value must be -inf (0xFF800000)
  if (val >= 0.0f) atomicMax((int*)addr, __float_as_int(val));
  else             atomicMin((unsigned int*)addr, __float_as_uint(val));
}

// ---------------------------------------------------------------------------
// One-shot f32 -> bf16 conversion of x (row-major [N][128]). Amortizes the
// RNE conversion out of the GEMM hot loop (8 waves/block were previously each
// re-converting the same A tile). 8 elements/thread, packed b128 store.
// ---------------------------------------------------------------------------
__global__ __launch_bounds__(256) void k_cvt_x(const float* __restrict__ x,
                                               unsigned short* __restrict__ xbf) {
  const size_t i = ((size_t)blockIdx.x * 256 + threadIdx.x) * 8;
  if (i >= (size_t)NN * DF) return;
  const v8f v = *(const v8f*)(x + i);
  union { unsigned short u16[8]; uint4 q; } o;
#pragma unroll
  for (int j = 0; j < 8; ++j) o.u16[j] = f2bf(v[j]);
  *(uint4*)(xbf + i) = o.q;
}

// ---------------------------------------------------------------------------
// Pack W_l / W_r (128x128 f32, row-major, K-major rows) into bf16 WMMA B
// fragments: layout [mat][kt(4)][nt(8)][lane(32)][elem(16)].
// B is KxN; lane L holds column n = L%16, kh = L/16; elem i -> k = kh*8+(i&7)+(i>>3)*16.
// ---------------------------------------------------------------------------
__global__ __launch_bounds__(256) void k_pack_w(const float* __restrict__ Wl,
                                                const float* __restrict__ Wr,
                                                unsigned short* __restrict__ pB) {
  const int idx = blockIdx.x * 256 + threadIdx.x;       // 2*4*8*32*16 = 32768
  if (idx >= 32768) return;
  const int i    = idx & 15;
  const int lane = (idx >> 4) & 31;
  const int nt   = (idx >> 9) & 7;
  const int kt   = (idx >> 12) & 3;
  const int mat  = idx >> 14;
  const int k = kt * 32 + (lane >> 4) * 8 + (i & 7) + ((i >> 3) << 4);
  const int n = nt * 16 + (lane & 15);
  const float* W = mat ? Wr : Wl;
  pB[idx] = f2bf(W[k * DF + n]);
}

// ---------------------------------------------------------------------------
// x_l = x@W_l + b_l ; x_r = x@W_r + b_r   via v_wmma_f32_16x16x32_bf16.
// One block = 8 waves = one 16-row tile; wave nt owns 16-col tile nt and
// reuses its A fragment for both weight matrices (2 WMMAs per K-step).
// A fragment comes straight from row-major bf16 x: 16-bit A 16x32 layout has
// lane (m=L%16, grp=L/16) holding K chunks {kb+grp*8..+7} and {kb+16+grp*8..+7}
// -> two contiguous b128 loads per K-step.
// ---------------------------------------------------------------------------
__global__ __launch_bounds__(256) void k_gemm(const unsigned short* __restrict__ xbf,
                                              const unsigned short* __restrict__ pB,
                                              const float* __restrict__ bl,
                                              const float* __restrict__ br,
                                              float* __restrict__ xl,
                                              float* __restrict__ xr) {
  const int lane = threadIdx.x & 31;
  const int nt   = threadIdx.x >> 5;    // col tile 0..7
  const int r0   = blockIdx.x << 4;     // 16-row tile (50000 = 3125*16)
  const int m    = lane & 15;
  const int grp  = lane >> 4;
  const unsigned short* xrow = xbf + (size_t)(r0 + m) * DF;
  const v16bf* pBl = (const v16bf*)pB;          // [mat][kt][nt][lane]
  const v16bf* pBr = pBl + 4 * 8 * 32;
  v8f cl = {}; v8f cr = {};
#pragma unroll
  for (int kt = 0; kt < 4; ++kt) {
    const int kb = kt * 32 + grp * 8;
    union { v16bf v; uint4 q[2]; } a;
    a.q[0] = *(const uint4*)(xrow + kb);        // K = kb .. kb+7
    a.q[1] = *(const uint4*)(xrow + kb + 16);   // K = kb+16 .. kb+23
    const int bo = (kt * 8 + nt) * 32 + lane;
    const v16bf bfl = pBl[bo];
    const v16bf bfr = pBr[bo];
    cl = __builtin_amdgcn_wmma_f32_16x16x32_bf16(false, a.v, false, bfl,
                                                 (short)0, cl, false, false);
    cr = __builtin_amdgcn_wmma_f32_16x16x32_bf16(false, a.v, false, bfr,
                                                 (short)0, cr, false, false);
  }
  // C/D layout: VGPR i -> row m = i + 8*grp, col n = lane%16
  const int n = (nt << 4) + m;
  const float addl = bl[n], addr2 = br[n];
#pragma unroll
  for (int i = 0; i < 8; ++i) {
    const int row = r0 + i + (grp << 3);
    xl[(size_t)row * DF + n] = cl[i] + addl;
    xr[(size_t)row * DF + n] = cr[i] + addr2;
  }
}

// ---------------------------------------------------------------------------
// Init: out accumulator = 0, segment max = -inf, segment sum = 0
// ---------------------------------------------------------------------------
__global__ __launch_bounds__(256) void k_init(float* __restrict__ out,
                                              float* __restrict__ mbuf,
                                              float* __restrict__ sbuf) {
  const size_t i = (size_t)blockIdx.x * 256 + threadIdx.x;
  if (i < (size_t)NN * DF) out[i] = 0.0f;
  if (i < (size_t)NN * HH) { mbuf[i] = __int_as_float(0xFF800000); sbuf[i] = 0.0f; }
}

// ---------------------------------------------------------------------------
// Per-edge GATv2 logits: one wave per edge; lane covers 4 channels; head h
// spans lanes [8h, 8h+8) -> xor-reduce within groups of 8.
// x_l / x_r gathers hit L2 (25.6 MB each, 192 MB L2).
// ---------------------------------------------------------------------------
__global__ __launch_bounds__(256) void k_logits(const int* __restrict__ src,
                                                const int* __restrict__ dst,
                                                const float* __restrict__ ew,
                                                const float* __restrict__ We,
                                                const float* __restrict__ att,
                                                const float* __restrict__ xl,
                                                const float* __restrict__ xr,
                                                float* __restrict__ logits) {
  const int lane = threadIdx.x & 31;
  const int e = blockIdx.x * 8 + (threadIdx.x >> 5);
  if (e >= EE) return;
  // stream-ahead prefetch of the index/weight arrays (global_prefetch_b8)
  if (e + 2048 < EE) {
    __builtin_prefetch(&src[e + 2048], 0, 0);
    __builtin_prefetch(&dst[e + 2048], 0, 0);
    __builtin_prefetch(&ew[e + 2048], 0, 0);
  }
  const int s = src[e], d = dst[e];
  const float w = ew[e];
  const int ch = lane * 4;
  const v4f xlv = *(const v4f*)(xl + (size_t)s * DF + ch);
  const v4f xrv = *(const v4f*)(xr + (size_t)d * DF + ch);
  const v4f wev = *(const v4f*)(We + ch);
  const v4f av  = *(const v4f*)(att + ch);
  float part = 0.0f;
#pragma unroll
  for (int i = 0; i < 4; ++i) {
    float z = xlv[i] + xrv[i] + w * wev[i];
    z = z >= 0.0f ? z : 0.2f * z;          // leaky_relu slope 0.2
    part += z * av[i];
  }
  part += __shfl_xor(part, 1, 32);
  part += __shfl_xor(part, 2, 32);
  part += __shfl_xor(part, 4, 32);
  if ((lane & 7) == 0) logits[(size_t)e * HH + (lane >> 3)] = part;
}

// ---------------------------------------------------------------------------
// Segment max over dst (L2-side float atomics via sign trick)
// ---------------------------------------------------------------------------
__global__ __launch_bounds__(256) void k_segmax(const int* __restrict__ dst,
                                                const float* __restrict__ logits,
                                                float* __restrict__ mbuf) {
  const size_t i = (size_t)blockIdx.x * 256 + threadIdx.x;
  if (i >= (size_t)EE * HH) return;
  const int e = (int)(i >> 2), h = (int)(i & 3);
  atomicMaxF(&mbuf[(size_t)dst[e] * HH + h], logits[i]);
}

// p = exp(logit - m[dst]); segment sum
__global__ __launch_bounds__(256) void k_expsum(const int* __restrict__ dst,
                                                const float* __restrict__ logits,
                                                const float* __restrict__ mbuf,
                                                float* __restrict__ p,
                                                float* __restrict__ sbuf) {
  const size_t i = (size_t)blockIdx.x * 256 + threadIdx.x;
  if (i >= (size_t)EE * HH) return;
  const int e = (int)(i >> 2), h = (int)(i & 3);
  const size_t di = (size_t)dst[e] * HH + h;
  const float pv = __expf(logits[i] - mbuf[di]);
  p[i] = pv;
  atomicAdd(&sbuf[di], pv);
}

// ---------------------------------------------------------------------------
// Weighted scatter: out[dst] += alpha * x_l[src]; wave per edge, 4 ch/lane.
// ---------------------------------------------------------------------------
__global__ __launch_bounds__(256) void k_scatter(const int* __restrict__ src,
                                                 const int* __restrict__ dst,
                                                 const float* __restrict__ p,
                                                 const float* __restrict__ sbuf,
                                                 const float* __restrict__ xl,
                                                 float* __restrict__ out) {
  const int lane = threadIdx.x & 31;
  const int e = blockIdx.x * 8 + (threadIdx.x >> 5);
  if (e >= EE) return;
  if (e + 2048 < EE) {
    __builtin_prefetch(&src[e + 2048], 0, 0);
    __builtin_prefetch(&dst[e + 2048], 0, 0);
  }
  const int s = src[e], d = dst[e];
  const int h = lane >> 3;
  const float alpha = p[(size_t)e * HH + h] / (sbuf[(size_t)d * HH + h] + 1e-16f);
  const int ch = lane * 4;
  const v4f xlv = *(const v4f*)(xl + (size_t)s * DF + ch);
  float* ob = out + (size_t)d * DF + ch;
  atomicAdd(ob + 0, alpha * xlv[0]);
  atomicAdd(ob + 1, alpha * xlv[1]);
  atomicAdd(ob + 2, alpha * xlv[2]);
  atomicAdd(ob + 3, alpha * xlv[3]);
}

// ---------------------------------------------------------------------------
// BatchNorm column statistics (bias folded in): one block per column.
// ---------------------------------------------------------------------------
__global__ __launch_bounds__(256) void k_bnstats(const float* __restrict__ out,
                                                 const float* __restrict__ bias,
                                                 float* __restrict__ musig) {
  __shared__ float ss[256], sq[256];
  const int c = blockIdx.x;
  const float b = bias[c];
  float s = 0.0f, q = 0.0f;
  for (int r = threadIdx.x; r < NN; r += 256) {
    const float v = out[(size_t)r * DF + c] + b;
    s += v; q += v * v;
  }
  ss[threadIdx.x] = s; sq[threadIdx.x] = q;
  __syncthreads();
  for (int o = 128; o > 0; o >>= 1) {
    if (threadIdx.x < o) {
      ss[threadIdx.x] += ss[threadIdx.x + o];
      sq[threadIdx.x] += sq[threadIdx.x + o];
    }
    __syncthreads();
  }
  if (threadIdx.x == 0) {
    const float mu  = ss[0] / (float)NN;
    const float var = sq[0] / (float)NN - mu * mu;
    musig[c]      = mu;
    musig[DF + c] = rsqrtf(var + 1e-5f);
  }
}

// Normalize + affine + leaky_relu(0.01), in place in d_out
__global__ __launch_bounds__(256) void k_bnact(float* __restrict__ out,
                                               const float* __restrict__ bias,
                                               const float* __restrict__ musig,
                                               const float* __restrict__ gamma,
                                               const float* __restrict__ beta) {
  const size_t i = (size_t)blockIdx.x * 256 + threadIdx.x;
  if (i >= (size_t)NN * DF) return;
  const int c = (int)(i & (DF - 1));
  float v = out[i] + bias[c];
  v = (v - musig[c]) * musig[DF + c] * gamma[c] + beta[c];
  out[i] = v >= 0.0f ? v : 0.01f * v;
}

extern "C" void kernel_launch(void* const* d_in, const int* in_sizes, int n_in,
                              void* d_out, int out_size, void* d_ws, size_t ws_size,
                              hipStream_t stream) {
  (void)in_sizes; (void)n_in; (void)out_size; (void)ws_size;
  const float* x    = (const float*)d_in[0];
  const int*   eidx = (const int*)d_in[1];       // [2][E]; row 0 = src, row 1 = dst
  const float* ew   = (const float*)d_in[2];
  const float* Wl   = (const float*)d_in[3];
  const float* bl   = (const float*)d_in[4];
  const float* Wr   = (const float*)d_in[5];
  const float* br   = (const float*)d_in[6];
  const float* We   = (const float*)d_in[7];
  const float* att  = (const float*)d_in[8];     // [H][C] flat == 128 channels
  const float* bias = (const float*)d_in[9];
  const float* gam  = (const float*)d_in[10];
  const float* bet  = (const float*)d_in[11];
  float* out = (float*)d_out;                    // [N][128]; also pre-BN accumulator

  const int* src = eidx;
  const int* dst = eidx + EE;

  char* ws = (char*)d_ws;
  size_t off = 0;
  auto alloc = [&](size_t bytes) -> void* {
    void* ptr = ws + off;
    off = (off + bytes + 255) & ~(size_t)255;
    return ptr;
  };
  float*          xl     = (float*)alloc((size_t)NN * DF * 4);
  float*          xr     = (float*)alloc((size_t)NN * DF * 4);
  unsigned short* xbf    = (unsigned short*)alloc((size_t)NN * DF * 2);
  unsigned short* pB     = (unsigned short*)alloc(2 * 4 * 8 * 32 * 16 * 2);
  float*          logits = (float*)alloc((size_t)EE * HH * 4);
  float*          pbuf   = (float*)alloc((size_t)EE * HH * 4);
  float*          mbuf   = (float*)alloc((size_t)NN * HH * 4);
  float*          sbuf   = (float*)alloc((size_t)NN * HH * 4);
  float*          musig  = (float*)alloc(2 * DF * 4);

  k_init<<<(NN * DF + 255) / 256, 256, 0, stream>>>(out, mbuf, sbuf);
  k_cvt_x<<<(NN * DF / 8 + 255) / 256, 256, 0, stream>>>(x, xbf);
  k_pack_w<<<32768 / 256, 256, 0, stream>>>(Wl, Wr, pB);
  k_gemm<<<NN / 16, 256, 0, stream>>>(xbf, pB, bl, br, xl, xr);
  k_logits<<<(EE + 7) / 8, 256, 0, stream>>>(src, dst, ew, We, att, xl, xr, logits);
  k_segmax<<<(EE * HH + 255) / 256, 256, 0, stream>>>(dst, logits, mbuf);
  k_expsum<<<(EE * HH + 255) / 256, 256, 0, stream>>>(dst, logits, mbuf, pbuf, sbuf);
  k_scatter<<<(EE + 7) / 8, 256, 0, stream>>>(src, dst, pbuf, sbuf, xl, out);
  k_bnstats<<<DF, 256, 0, stream>>>(out, bias, musig);
  k_bnact<<<(NN * DF + 255) / 256, 256, 0, stream>>>(out, bias, musig, gam, bet);
}
